// GNNLSTM_50766513439454
// MI455X (gfx1250) — matrix operand order repaired and checked by reading, not compiled, not run on hardware
//
#include <hip/hip_runtime.h>

typedef __attribute__((ext_vector_type(16))) __bf16 v16bf;
typedef __attribute__((ext_vector_type(8)))  float  v8f;

#define kH 512          // LSTM hidden
#define kS 128          // sequence length
#define kN 16           // nodes per graph (one WMMA M tile)
#define kG 256          // layer-2 output features
#define kOutStride 259  // G + hideout(2) + timestep(1)

// ---------------------------------------------------------------------------
// bf16 helpers (storage kept as ushort so host code never touches __bf16)
// ---------------------------------------------------------------------------
__device__ __forceinline__ unsigned short f2bf(float f) {
  unsigned u = __float_as_uint(f);
  unsigned r = (u + 0x7FFFu + ((u >> 16) & 1u)) >> 16;  // round-nearest-even
  return (unsigned short)r;
}
__device__ __forceinline__ float bf2f(unsigned short b) {
  return __uint_as_float(((unsigned)b) << 16);
}
__device__ __forceinline__ float sigmoid_(float x) { return 1.0f / (1.0f + __expf(-x)); }

// Position of A-matrix element (row M, k = K%32) inside one 16x32 bf16 A tile
// stored as [lane 0..31][slot 0..15] matching the V_WMMA 16-bit A layout:
//   lanes 0-15: M=lane, K = slot (slots 0-7) / slot+8 (slots 8-15)
//   lanes 16-31: M=lane-16, K = slot+8 / slot+16
__device__ __forceinline__ int apos(int M, int k) {
  int lane = M + (((k >> 3) & 1) << 4);
  int slot = (k & 7) | ((k & 16) >> 1);
  return lane * 16 + slot;
}

__device__ __forceinline__ v8f splat8(float x) {
  v8f v;
#pragma unroll
  for (int i = 0; i < 8; ++i) v[i] = x;
  return v;
}

__device__ __forceinline__ v8f wmma_bf16(v16bf a, v16bf b, v8f c) {
  // D = A(16x32 bf16) * B(32x16 bf16) + C(f32)
  return __builtin_amdgcn_wmma_f32_16x16x32_bf16(false, a, false, b, (short)0, c,
                                                 false, false);
}

// Packed B operand: tiles [nt][kt][lane 0..31][slot 0..15], each lane's 16
// bf16 contiguous (32B) so the load is two global_load_b128 per lane.
__device__ __forceinline__ v16bf ld_b(const unsigned short* __restrict__ p,
                                      int nt, int kt, int KT, int L) {
  return *(const v16bf*)(p + ((size_t)(nt * KT + kt) * 32 + L) * 16);
}
// A operand from LDS: [kt][lane][slot]
__device__ __forceinline__ v16bf ld_a(const unsigned short* p, int kt, int L) {
  return *(const v16bf*)(p + (kt * 32 + L) * 16);
}

// ---------------------------------------------------------------------------
// Pack fp32 weight W[NN][KK] (row n = output/gate column of x@W.T) into bf16
// WMMA B-tile layout.  B element (K,Ncol): lane = (N%16) + 16*((K>>3)&1),
// slot = (K&7) | ((K&16)>>1).  K beyond KK (W_ih pad 16->32) is zero-filled.
// ---------------------------------------------------------------------------
__global__ void pack_weight(const float* __restrict__ W,
                            unsigned short* __restrict__ dst,
                            int NN, int KK, int KT, int total) {
  int idx = blockIdx.x * blockDim.x + threadIdx.x;
  if (idx >= total) return;
  int slot = idx & 15;
  int lane = (idx >> 4) & 31;
  int kt   = (idx >> 9) % KT;
  int nt   = (idx >> 9) / KT;
  int n = nt * 16 + (lane & 15);
  int k = kt * 32 + (lane >> 4) * 8 + (slot & 7) + (slot >> 3) * 16;
  float v = (k < KK) ? W[(size_t)n * KK + k] : 0.0f;
  dst[idx] = f2bf(v);
}

// Exclusive max over the 16 nodes per feature column (SAGE pool aggregator,
// fully-connected graph without self loops). Reads plain [node][col] bf16,
// writes the result directly into WMMA-A layout.
__device__ __forceinline__ void excl_max_to_neighA(const unsigned short* Mb,
                                                   unsigned short* Na, int tid) {
  for (int c = tid; c < kH; c += 256) {
    float m0 = -3.402823466e38f, m1 = m0;
    int am = 0;
#pragma unroll
    for (int i = 0; i < 16; ++i) {
      float v = bf2f(Mb[i * kH + c]);
      if (v > m0) { m1 = m0; m0 = v; am = i; }
      else if (v > m1) { m1 = v; }
    }
    unsigned short bmax = f2bf(m0), bsec = f2bf(m1);
    int base = (c >> 5) * 512, kk = c & 31;
#pragma unroll
    for (int i = 0; i < 16; ++i)
      Na[base + apos(i, kk)] = (i == am) ? bsec : bmax;
  }
}

// ---------------------------------------------------------------------------
// Fused LSTM (128 steps) + 2x SAGEConv(pool) + mean + concat.
// One workgroup (8 wave32) per batch element: its 16 node-sequences are one
// WMMA M tile; h lives in LDS in A-operand layout for the whole recurrence.
// Wave w owns h-columns [64w, 64w+64): 4 col-tiles, each needing the i/f/g/o
// gate tiles (same C-layout element mapping -> register-resident cell update).
// __launch_bounds__(256, 2): 8 waves/block = 2 waves/SIMD on a WGP, so force
// the allocator to fit two waves per SIMD (also kills most s_set_vgpr_msb).
// ---------------------------------------------------------------------------
__global__ __launch_bounds__(256, 2) void gnn_lstm_fused(
    const float* __restrict__ agent_obs,   // [B,S,N,D]
    const float* __restrict__ hideout,     // [B,2]
    const float* __restrict__ timestep,    // [B,1]
    const float* __restrict__ b_ih,        // [4H]
    const float* __restrict__ b_hh,        // [4H]
    const float* __restrict__ pool_b1,     // [H]
    const float* __restrict__ bias1,       // [H]
    const float* __restrict__ pool_b2,     // [H]
    const float* __restrict__ bias2,       // [G]
    const unsigned short* __restrict__ whh,  // packed bf16 [128nt][16kt]
    const unsigned short* __restrict__ wih,  // packed bf16 [128nt][1kt]
    const unsigned short* __restrict__ pw1,  // [32nt][16kt]
    const unsigned short* __restrict__ sw1,
    const unsigned short* __restrict__ nw1,
    const unsigned short* __restrict__ pw2,
    const unsigned short* __restrict__ sw2,  // [16nt][16kt]
    const unsigned short* __restrict__ nw2,
    float* __restrict__ out)               // [B, 259]
{
  __shared__ __align__(64) unsigned short Hbuf[16 * 512];  // h / r, A layout (16KB)
  __shared__ __align__(64) unsigned short Xtile[512];      // x_t A tile, K padded
  __shared__ __align__(64) unsigned short Mbuf[16 * 512];  // relu(pool) [node][col]
  __shared__ __align__(64) unsigned short NeighA[16 * 512];// neigh, A layout
  __shared__ float Colsum[kG];

  const int b    = blockIdx.x;
  const int tid  = threadIdx.x;
  const int w    = tid >> 5;   // wave id 0..7
  const int L    = tid & 31;   // lane
  const int ln   = L & 15;     // N index inside tile
  const int half = L >> 4;     // 0: rows M=0..7, 1: rows M=8..15

  // ---- zero init (h0 = 0, padded K lanes of Xtile = 0) ----
  for (int i = tid; i < 16 * 512; i += 256) Hbuf[i] = 0;
  for (int i = tid; i < 512; i += 256) Xtile[i] = 0;
  __syncthreads();

  // this thread's x element: node xn, feature xd
  const int xn = tid >> 4, xd = tid & 15;
  const int xpos = apos(xn, xd);
  const float* xbase = agent_obs + (size_t)b * kS * 256 + xn * 16 + xd;

  // combined gate biases for this wave's 4 col-tiles
  float bi[4], bff[4], bg[4], bo[4];
#pragma unroll
  for (int t = 0; t < 4; ++t) {
    int hc = 64 * w + 16 * t + ln;
    bi[t]  = b_ih[hc]        + b_hh[hc];
    bff[t] = b_ih[512 + hc]  + b_hh[512 + hc];
    bg[t]  = b_ih[1024 + hc] + b_hh[1024 + hc];
    bo[t]  = b_ih[1536 + hc] + b_hh[1536 + hc];
  }

  v8f cst[4];  // register-resident cell state c for (M = v+8*half, col)
#pragma unroll
  for (int t = 0; t < 4; ++t) cst[t] = splat8(0.0f);

  // =================== LSTM recurrence ===================
  for (int s = 0; s < kS; ++s) {
    Xtile[xpos] = f2bf(xbase[s * 256]);  // stage x_t into A-layout tile
    __syncthreads();                     // Xtile + Hbuf(h_{s}) ready

    v16bf ax = *(const v16bf*)&Xtile[L * 16];
    v8f hstl[4];
#pragma unroll
    for (int t = 0; t < 4; ++t) {
      const int ct = 4 * w + t;  // col tile within H (0..31)
      v8f ai = splat8(bi[t]), af = splat8(bff[t]);
      v8f ag = splat8(bg[t]), ao = splat8(bo[t]);
      // x @ W_ih.T  (K padded 16->32 with zeros)
      ai = wmma_bf16(ax, ld_b(wih, ct,      0, 1, L), ai);
      af = wmma_bf16(ax, ld_b(wih, 32 + ct, 0, 1, L), af);
      ag = wmma_bf16(ax, ld_b(wih, 64 + ct, 0, 1, L), ag);
      ao = wmma_bf16(ax, ld_b(wih, 96 + ct, 0, 1, L), ao);
      // h @ W_hh.T : K = 512 = 16 bf16 k-tiles, A shared by all 4 gates.
      // unroll 2: pairs the LDS A loads for ILP without the x4 register blowup.
#pragma unroll 2
      for (int kt = 0; kt < 16; ++kt) {
        v16bf a = ld_a(Hbuf, kt, L);
        ai = wmma_bf16(a, ld_b(whh, ct,      kt, 16, L), ai);
        af = wmma_bf16(a, ld_b(whh, 32 + ct, kt, 16, L), af);
        ag = wmma_bf16(a, ld_b(whh, 64 + ct, kt, 16, L), ag);
        ao = wmma_bf16(a, ld_b(whh, 96 + ct, kt, 16, L), ao);
      }
      // cell update: acc element (v,lane) is (M=v+8*half, col) for all gates
#pragma unroll
      for (int v = 0; v < 8; ++v) {
        float ig = sigmoid_(ai[v]);
        float fg = sigmoid_(af[v]);
        float gg = tanhf(ag[v]);
        float og = sigmoid_(ao[v]);
        float c  = fg * cst[t][v] + ig * gg;
        cst[t][v] = c;
        hstl[t][v] = og * tanhf(c);
      }
    }
    __syncthreads();  // all reads of Hbuf(h_s) complete -> safe to overwrite
#pragma unroll
    for (int t = 0; t < 4; ++t) {
      int colK = 64 * w + 16 * t + ln;
      int base = (colK >> 5) * 512, kk = colK & 31;
#pragma unroll
      for (int v = 0; v < 8; ++v)
        Hbuf[base + apos(v + 8 * half, kk)] = f2bf(hstl[t][v]);
    }
  }
  __syncthreads();  // Hbuf = hn (A layout)

  // =================== SAGE layer 1 ===================
  // m = relu(hn @ pool_W1.T + pool_b1)
  for (int j = 0; j < 4; ++j) {
    int nt = 4 * w + j;
    v8f acc = splat8(pool_b1[nt * 16 + ln]);
#pragma unroll 2
    for (int kt = 0; kt < 16; ++kt)
      acc = wmma_bf16(ld_a(Hbuf, kt, L), ld_b(pw1, nt, kt, 16, L), acc);
#pragma unroll
    for (int v = 0; v < 8; ++v)
      Mbuf[(v + 8 * half) * kH + nt * 16 + ln] = f2bf(fmaxf(acc[v], 0.0f));
  }
  __syncthreads();
  excl_max_to_neighA(Mbuf, NeighA, tid);
  __syncthreads();

  // r = tanh(hn @ self_W1.T + neigh @ neigh_W1.T + bias1)
  v8f r1[4];
  for (int j = 0; j < 4; ++j) {
    int nt = 4 * w + j;
    v8f acc = splat8(bias1[nt * 16 + ln]);
#pragma unroll 2
    for (int kt = 0; kt < 16; ++kt)
      acc = wmma_bf16(ld_a(Hbuf, kt, L), ld_b(sw1, nt, kt, 16, L), acc);
#pragma unroll 2
    for (int kt = 0; kt < 16; ++kt)
      acc = wmma_bf16(ld_a(NeighA, kt, L), ld_b(nw1, nt, kt, 16, L), acc);
#pragma unroll
    for (int v = 0; v < 8; ++v) r1[j][v] = tanhf(acc[v]);
  }
  __syncthreads();  // hn reads done -> overwrite Hbuf with r
#pragma unroll
  for (int j = 0; j < 4; ++j) {
    int col = (4 * w + j) * 16 + ln;
    int base = (col >> 5) * 512, kk = col & 31;
#pragma unroll
    for (int v = 0; v < 8; ++v)
      Hbuf[base + apos(v + 8 * half, kk)] = f2bf(r1[j][v]);
  }
  Colsum[tid & (kG - 1)] = 0.0f;
  __syncthreads();

  // =================== SAGE layer 2 ===================
  for (int j = 0; j < 4; ++j) {
    int nt = 4 * w + j;
    v8f acc = splat8(pool_b2[nt * 16 + ln]);
#pragma unroll 2
    for (int kt = 0; kt < 16; ++kt)
      acc = wmma_bf16(ld_a(Hbuf, kt, L), ld_b(pw2, nt, kt, 16, L), acc);
#pragma unroll
    for (int v = 0; v < 8; ++v)
      Mbuf[(v + 8 * half) * kH + nt * 16 + ln] = f2bf(fmaxf(acc[v], 0.0f));
  }
  __syncthreads();
  excl_max_to_neighA(Mbuf, NeighA, tid);
  __syncthreads();

  // out2 = r @ self_W2.T + neigh2 @ neigh_W2.T + bias2 ; mean over nodes
  for (int j = 0; j < 2; ++j) {
    int nt = 2 * w + j;  // 16 output tiles (G = 256)
    v8f acc = splat8(bias2[nt * 16 + ln]);
#pragma unroll 2
    for (int kt = 0; kt < 16; ++kt)
      acc = wmma_bf16(ld_a(Hbuf, kt, L), ld_b(sw2, nt, kt, 16, L), acc);
#pragma unroll 2
    for (int kt = 0; kt < 16; ++kt)
      acc = wmma_bf16(ld_a(NeighA, kt, L), ld_b(nw2, nt, kt, 16, L), acc);
    float ps = 0.0f;
#pragma unroll
    for (int v = 0; v < 8; ++v) ps += acc[v];   // partial sum over 8 rows
    atomicAdd(&Colsum[nt * 16 + ln], ps);       // ds_add_f32, both lane halves
  }
  __syncthreads();

  // mean + concat [r(256), hideout(2), timestep(1)]
  out[(size_t)b * kOutStride + tid] = Colsum[tid & (kG - 1)] * (1.0f / 16.0f);
  if (tid == 0) {
    out[(size_t)b * kOutStride + 256] = hideout[2 * b];
    out[(size_t)b * kOutStride + 257] = hideout[2 * b + 1];
    out[(size_t)b * kOutStride + 258] = timestep[b];
  }
}

// ---------------------------------------------------------------------------
extern "C" void kernel_launch(void* const* d_in, const int* in_sizes, int n_in,
                              void* d_out, int out_size, void* d_ws, size_t ws_size,
                              hipStream_t stream) {
  (void)in_sizes; (void)n_in; (void)out_size; (void)ws_size;
  const float* agent_obs = (const float*)d_in[0];
  const float* hideout   = (const float*)d_in[1];
  const float* timestep  = (const float*)d_in[2];
  // d_in[3] = num_agents (constant 16, unused)
  const float* W_ih      = (const float*)d_in[4];
  const float* W_hh      = (const float*)d_in[5];
  const float* b_ih      = (const float*)d_in[6];
  const float* b_hh      = (const float*)d_in[7];
  const float* pool_W1   = (const float*)d_in[8];
  const float* pool_b1   = (const float*)d_in[9];
  const float* self_W1   = (const float*)d_in[10];
  const float* neigh_W1  = (const float*)d_in[11];
  const float* bias1     = (const float*)d_in[12];
  const float* pool_W2   = (const float*)d_in[13];
  const float* pool_b2   = (const float*)d_in[14];
  const float* self_W2   = (const float*)d_in[15];
  const float* neigh_W2  = (const float*)d_in[16];
  const float* bias2     = (const float*)d_in[17];
  float* out = (float*)d_out;

  // Workspace layout (bf16 stored as ushort), ~4.63 MiB total.
  unsigned short* ws  = (unsigned short*)d_ws;
  unsigned short* whh = ws;                  // 128nt*16kt*512 = 1048576
  unsigned short* wih = whh + 1048576;       // 128nt* 1kt*512 =   65536
  unsigned short* pw1 = wih + 65536;         //  32nt*16kt*512 =  262144
  unsigned short* sw1 = pw1 + 262144;
  unsigned short* nw1 = sw1 + 262144;
  unsigned short* pw2 = nw1 + 262144;
  unsigned short* sw2 = pw2 + 262144;        //  16nt*16kt*512 =  131072
  unsigned short* nw2 = sw2 + 131072;

  const int T = 256;
  pack_weight<<<(1048576 + T - 1) / T, T, 0, stream>>>(W_hh, whh, 2048, 512, 16, 1048576);
  pack_weight<<<(65536   + T - 1) / T, T, 0, stream>>>(W_ih, wih, 2048, 16,  1,  65536);
  pack_weight<<<(262144  + T - 1) / T, T, 0, stream>>>(pool_W1,  pw1, 512, 512, 16, 262144);
  pack_weight<<<(262144  + T - 1) / T, T, 0, stream>>>(self_W1,  sw1, 512, 512, 16, 262144);
  pack_weight<<<(262144  + T - 1) / T, T, 0, stream>>>(neigh_W1, nw1, 512, 512, 16, 262144);
  pack_weight<<<(262144  + T - 1) / T, T, 0, stream>>>(pool_W2,  pw2, 512, 512, 16, 262144);
  pack_weight<<<(131072  + T - 1) / T, T, 0, stream>>>(self_W2,  sw2, 256, 512, 16, 131072);
  pack_weight<<<(131072  + T - 1) / T, T, 0, stream>>>(neigh_W2, nw2, 256, 512, 16, 131072);

  gnn_lstm_fused<<<32, 256, 0, stream>>>(
      agent_obs, hideout, timestep, b_ih, b_hh,
      pool_b1, bias1, pool_b2, bias2,
      whh, wih, pw1, sw1, nw1, pw2, sw2, nw2, out);
}
